// DeepseekV4MoE_67637144978442
// MI455X (gfx1250) — matrix operand order repaired and checked by reading, not compiled, not run on hardware
//
#include <hip/hip_runtime.h>

typedef _Float16 half_t;
typedef __attribute__((ext_vector_type(16))) _Float16 v16h;
typedef __attribute__((ext_vector_type(8)))  float    v8f;

constexpr int kT   = 2048;   // tokens
constexpr int kE   = 32;     // experts
constexpr int kK   = 6;      // top-k
constexpr int kH   = 2048;   // hidden
constexpr int kI   = 1024;   // intermediate
constexpr int kCap = 768;    // capacity
constexpr float kScale = 1.5f;
constexpr float kLimit = 10.0f;

// LDS row stride 20 uints: rows stay 16B-aligned for ds_store_b128,
// and row*20 mod 64 covers 16 distinct banks across a half-wave.
constexpr int BM = 128, BN = 64, BK = 32, PAD = 20;

__device__ __forceinline__ v8f wmma16(v16h a, v16h b, v8f c) {
  // D = A(16x32 f16) x B(32x16 f16) + C(16x16 f32)
  return __builtin_amdgcn_wmma_f32_16x16x32_f16(false, a, false, b, (short)0, c, false, false);
}

union FragU { v16h v; unsigned int u[8]; };
union PackU { unsigned int u32; half_t h[2]; };

__device__ __forceinline__ unsigned int pk2(float a, float b) {
  PackU p; p.h[0] = (half_t)a; p.h[1] = (half_t)b; return p.u32;
}

// fast silu: x * rcp(1+exp(-x)); v_rcp_f32 approx (~1ulp) is plenty for an f16 result
__device__ __forceinline__ float fast_silu(float x) {
  return x * __builtin_amdgcn_rcpf(1.f + __expf(-x));
}

// ---------------- 1) convert x -> f16, zero expert counts ----------------
__global__ __launch_bounds__(256) void prep_kernel(const float* __restrict__ x,
                                                   half_t* __restrict__ h16,
                                                   int* __restrict__ counts) {
  size_t base = (size_t)blockIdx.x * 2048 + threadIdx.x;
#pragma unroll
  for (int j = 0; j < 8; ++j) {
    size_t idx = base + (size_t)j * 256;
    h16[idx] = (half_t)x[idx];
  }
  if (blockIdx.x == 0 && threadIdx.x < kE) counts[threadIdx.x] = 0;
}

// ---------------- 2) router: sqrt(softplus) scores + biased top-6 ----------------
__global__ __launch_bounds__(256) void router_kernel(const float* __restrict__ x,
                                                     const float* __restrict__ gw,
                                                     const float* __restrict__ bias,
                                                     int* __restrict__ topk_idx,
                                                     float* __restrict__ topk_w) {
  const int wv = threadIdx.x >> 5, lane = threadIdx.x & 31;
  const int t = blockIdx.x * 8 + wv;                 // one wave per token, one lane per expert
  const float4* hrow = (const float4*)(x  + (size_t)t    * kH);
  const float4* grow = (const float4*)(gw + (size_t)lane * kH);
  float acc = 0.f;
  for (int k = 0; k < kH / 4; ++k) {
    float4 hv = hrow[k], gv = grow[k];
    acc += hv.x * gv.x + hv.y * gv.y + hv.z * gv.z + hv.w * gv.w;
  }
  float sp    = (acc > 20.f) ? acc : log1pf(__expf(acc));
  float score = sqrtf(sp);
  float b     = score + bias[lane];

  float ssum = 0.f;
  int   wi[kK];
  float wsc[kK];
#pragma unroll
  for (int r = 0; r < kK; ++r) {
    float v = b; int idx = lane;
#pragma unroll
    for (int off = 16; off >= 1; off >>= 1) {       // wave32 argmax (ties -> lowest index)
      float ov = __shfl_xor(v, off, 32);
      int   oi = __shfl_xor(idx, off, 32);
      if (ov > v || (ov == v && oi < idx)) { v = ov; idx = oi; }
    }
    float sc = __shfl(score, idx, 32);              // unbiased score of winner
    wi[r] = idx; wsc[r] = sc; ssum += sc;
    if (lane == idx) b = -INFINITY;
  }
  if (lane == 0) {
    float inv = kScale / ssum;
#pragma unroll
    for (int r = 0; r < kK; ++r) {
      topk_idx[t * kK + r] = wi[r];
      topk_w  [t * kK + r] = wsc[r] * inv;
    }
  }
}

// ---------------- 3) capacity dispatch ----------------
__global__ __launch_bounds__(256) void dispatch_kernel(const int* __restrict__ topk_idx,
                                                       int* __restrict__ counts,
                                                       int* __restrict__ slot_token,
                                                       int* __restrict__ slot_pos) {
  int i = blockIdx.x * 256 + threadIdx.x;
  if (i >= kT * kK) return;
  int e   = topk_idx[i];
  int pos = atomicAdd(&counts[e], 1);
  slot_pos[i] = pos;
  if (pos < kCap) slot_token[e * kCap + pos] = i / kK;
}

// ---------------- 4/6) gate+up GEMM + SwiGLU (shared kernel; sTok==null => identity rows) ----
__global__ __launch_bounds__(256)
void moe_gateup_kernel(const half_t* __restrict__ A, const int* __restrict__ sTok,
                       const int* __restrict__ counts, const float* __restrict__ Wg,
                       const float* __restrict__ Wu, half_t* __restrict__ Out,
                       int M, int N, int Kd, float clampLim) {
  __shared__ unsigned int Alds[BM * PAD];
  __shared__ unsigned int Bgl [BN * PAD];
  __shared__ unsigned int Bul [BN * PAD];

  const int e   = blockIdx.z;
  int cnt       = counts ? counts[e] : M;
  if (cnt > M) cnt = M;
  const int m0  = blockIdx.x * BM;
  if (m0 >= cnt) return;                                   // uniform early exit
  const int n0  = blockIdx.y * BN;
  const int tid = threadIdx.x;

  // --- A: one gathered row per 2 threads, two b128 loads per K-step ---
  const int arow = tid >> 1;         // 0..127
  const int ah   = tid & 1;          // which 16-half half of the row
  {
  }
  int gr  = m0 + arow;
  int tok = sTok ? ((gr < cnt) ? sTok[e * kCap + gr] : 0) : gr;
  const uint4* Au4 = (const uint4*)A;                      // 8 halves per uint4
  const size_t abase = (size_t)tok * (size_t)(Kd >> 3) + (size_t)(ah * 2);

  // --- B: one row per 4 threads, two float4 loads per matrix per K-step ---
  const int brow = tid >> 2;         // 0..63
  const int bq   = tid & 3;          // quarter of the row
  const float4* Wg4 = (const float4*)Wg;
  const float4* Wu4 = (const float4*)Wu;
  const size_t wOff4 = ((size_t)e * (size_t)N * (size_t)Kd) >> 2;
  const size_t bbase = wOff4 + (size_t)(n0 + brow) * (size_t)(Kd >> 2) + (size_t)(bq * 2);

  const int wv = tid >> 5, lane = tid & 31;
  const int wm = (wv >> 1) * 32, wn = (wv & 1) * 32;
  const int lr = lane & 15, hi = lane >> 4;

  v8f cg[2][2] = {{{0},{0}},{{0},{0}}};
  v8f cu[2][2] = {{{0},{0}},{{0},{0}}};

  for (int kp = 0; kp < Kd; kp += BK) {
    const int ko8 = kp >> 3;   // uint4 step within A row
    const int ko4 = kp >> 2;   // float4 step within B row
    // issue all tile loads first (batched, one staggered loadcnt wait)
    uint4  a0v = Au4[abase + ko8];
    uint4  a1v = Au4[abase + ko8 + 1];
    float4 g0  = Wg4[bbase + ko4];
    float4 g1  = Wg4[bbase + ko4 + 1];
    float4 u0  = Wu4[bbase + ko4];
    float4 u1  = Wu4[bbase + ko4 + 1];
    if (kp + BK < Kd) {        // keep the HBM weight stream ahead (global_prefetch_b8)
      __builtin_prefetch(&Wg4[bbase + ko4 + (BK >> 2)], 0, 0);
      __builtin_prefetch(&Wu4[bbase + ko4 + (BK >> 2)], 0, 0);
    }
    uint4 pg, pu;
    pg.x = pk2(g0.x, g0.y); pg.y = pk2(g0.z, g0.w);
    pg.z = pk2(g1.x, g1.y); pg.w = pk2(g1.z, g1.w);
    pu.x = pk2(u0.x, u0.y); pu.y = pk2(u0.z, u0.w);
    pu.z = pk2(u1.x, u1.y); pu.w = pk2(u1.z, u1.w);
    *(uint4*)&Alds[arow * PAD + ah * 8]     = a0v;
    *(uint4*)&Alds[arow * PAD + ah * 8 + 4] = a1v;
    *(uint4*)&Bgl [brow * PAD + bq * 4]     = pg;
    *(uint4*)&Bul [brow * PAD + bq * 4]     = pu;
    __syncthreads();

    FragU a0, a1, bg0, bg1, bu0, bu1;
#pragma unroll
    for (int v = 0; v < 8; ++v) {
      int ca = v + (v & 4) + (hi << 2);   // ISA 16-bit A 16x32 layout
      int cb = v + (v & 4) + (hi << 2);   // ISA 16-bit B 32x16 layout
      a0.u[v]  = Alds[(wm +      lr) * PAD + ca];
      a1.u[v]  = Alds[(wm + 16 + lr) * PAD + ca];
      bg0.u[v] = Bgl [(wn +      lr) * PAD + cb];
      bg1.u[v] = Bgl [(wn + 16 + lr) * PAD + cb];
      bu0.u[v] = Bul [(wn +      lr) * PAD + cb];
      bu1.u[v] = Bul [(wn + 16 + lr) * PAD + cb];
    }
    cg[0][0] = wmma16(a0.v, bg0.v, cg[0][0]);
    cg[0][1] = wmma16(a0.v, bg1.v, cg[0][1]);
    cg[1][0] = wmma16(a1.v, bg0.v, cg[1][0]);
    cg[1][1] = wmma16(a1.v, bg1.v, cg[1][1]);
    cu[0][0] = wmma16(a0.v, bu0.v, cu[0][0]);
    cu[0][1] = wmma16(a0.v, bu1.v, cu[0][1]);
    cu[1][0] = wmma16(a1.v, bu0.v, cu[1][0]);
    cu[1][1] = wmma16(a1.v, bu1.v, cu[1][1]);
    __syncthreads();
  }

  half_t* out = Out + (size_t)e * (size_t)M * (size_t)N;
#pragma unroll
  for (int mi = 0; mi < 2; ++mi)
#pragma unroll
    for (int ni = 0; ni < 2; ++ni) {
      int n = n0 + wn + ni * 16 + lr;
#pragma unroll
      for (int r = 0; r < 8; ++r) {
        int m = m0 + wm + mi * 16 + hi * 8 + r;
        if (m < cnt) {
          float sv = fast_silu(cg[mi][ni][r]) * cu[mi][ni][r];   // silu(g)*u
          sv = fminf(fmaxf(sv, -clampLim), clampLim);
          out[(size_t)m * N + n] = (half_t)sv;
        }
      }
    }
}

// ---------------- 5/7) down-projection GEMM ----------------
__global__ __launch_bounds__(256)
void moe_down_kernel(const half_t* __restrict__ A, const int* __restrict__ counts,
                     const float* __restrict__ W, half_t* __restrict__ Out,
                     int M, int N, int Kd) {
  __shared__ unsigned int Alds[BM * PAD];
  __shared__ unsigned int Bl  [BN * PAD];

  const int e   = blockIdx.z;
  int cnt       = counts ? counts[e] : M;
  if (cnt > M) cnt = M;
  const int m0  = blockIdx.x * BM;
  if (m0 >= cnt) return;
  const int n0  = blockIdx.y * BN;
  const int tid = threadIdx.x;

  const int arow = tid >> 1;
  const int ah   = tid & 1;
  const uint4* Au4 = (const uint4*)A;
  const size_t abase = ((size_t)e * (size_t)M + (size_t)(m0 + arow)) * (size_t)(Kd >> 3)
                     + (size_t)(ah * 2);

  const int brow = tid >> 2;
  const int bq   = tid & 3;
  const float4* W4 = (const float4*)W;
  const size_t wOff4 = ((size_t)e * (size_t)N * (size_t)Kd) >> 2;
  const size_t bbase = wOff4 + (size_t)(n0 + brow) * (size_t)(Kd >> 2) + (size_t)(bq * 2);

  const int wv = tid >> 5, lane = tid & 31;
  const int wm = (wv >> 1) * 32, wn = (wv & 1) * 32;
  const int lr = lane & 15, hi = lane >> 4;

  v8f cc[2][2] = {{{0},{0}},{{0},{0}}};

  for (int kp = 0; kp < Kd; kp += BK) {
    const int ko8 = kp >> 3;
    const int ko4 = kp >> 2;
    uint4  a0v = Au4[abase + ko8];
    uint4  a1v = Au4[abase + ko8 + 1];
    float4 w0  = W4[bbase + ko4];
    float4 w1  = W4[bbase + ko4 + 1];
    if (kp + BK < Kd)
      __builtin_prefetch(&W4[bbase + ko4 + (BK >> 2)], 0, 0);
    uint4 pw;
    pw.x = pk2(w0.x, w0.y); pw.y = pk2(w0.z, w0.w);
    pw.z = pk2(w1.x, w1.y); pw.w = pk2(w1.z, w1.w);
    *(uint4*)&Alds[arow * PAD + ah * 8]     = a0v;
    *(uint4*)&Alds[arow * PAD + ah * 8 + 4] = a1v;
    *(uint4*)&Bl  [brow * PAD + bq * 4]     = pw;
    __syncthreads();

    FragU a0, a1, b0, b1;
#pragma unroll
    for (int v = 0; v < 8; ++v) {
      int ca = v + (v & 4) + (hi << 2);
      int cb = v + (v & 4) + (hi << 2);
      a0.u[v] = Alds[(wm +      lr) * PAD + ca];
      a1.u[v] = Alds[(wm + 16 + lr) * PAD + ca];
      b0.u[v] = Bl  [(wn +      lr) * PAD + cb];
      b1.u[v] = Bl  [(wn + 16 + lr) * PAD + cb];
    }
    cc[0][0] = wmma16(a0.v, b0.v, cc[0][0]);
    cc[0][1] = wmma16(a0.v, b1.v, cc[0][1]);
    cc[1][0] = wmma16(a1.v, b0.v, cc[1][0]);
    cc[1][1] = wmma16(a1.v, b1.v, cc[1][1]);
    __syncthreads();
  }

  half_t* out = Out + (size_t)e * (size_t)M * (size_t)N;
#pragma unroll
  for (int mi = 0; mi < 2; ++mi)
#pragma unroll
    for (int ni = 0; ni < 2; ++ni) {
      int n = n0 + wn + ni * 16 + lr;
#pragma unroll
      for (int r = 0; r < 8; ++r) {
        int m = m0 + wm + mi * 16 + hi * 8 + r;
        if (m < cnt) out[(size_t)m * N + n] = (half_t)cc[mi][ni][r];
      }
    }
}

// ---------------- 8) combine: routed gather + shared ----------------
__global__ __launch_bounds__(256)
void combine_kernel(const int* __restrict__ topk_idx, const float* __restrict__ topk_w,
                    const int* __restrict__ slot_pos, const half_t* __restrict__ yb,
                    const half_t* __restrict__ ysh, float* __restrict__ out) {
  const int t = blockIdx.x;
  const half_t* rows[kK];
  float w[kK];
#pragma unroll
  for (int k = 0; k < kK; ++k) {
    int ii = t * kK + k;
    int e = topk_idx[ii];
    int p = slot_pos[ii];
    bool ok = (p < kCap);
    rows[k] = ok ? (yb + ((size_t)e * kCap + p) * kH) : ysh;   // dummy ptr when dropped
    w[k]    = ok ? topk_w[ii] : 0.f;
  }
  for (int h = threadIdx.x; h < kH; h += 256) {
    float acc = (float)ysh[(size_t)t * kH + h];
#pragma unroll
    for (int k = 0; k < kK; ++k) acc += w[k] * (float)rows[k][h];
    out[(size_t)t * kH + h] = acc;
  }
}

// ---------------- host launcher ----------------
extern "C" void kernel_launch(void* const* d_in, const int* in_sizes, int n_in,
                              void* d_out, int out_size, void* d_ws, size_t ws_size,
                              hipStream_t stream) {
  (void)in_sizes; (void)n_in; (void)out_size; (void)ws_size;
  const float* x      = (const float*)d_in[0];
  const float* gate_w = (const float*)d_in[1];
  const float* bias   = (const float*)d_in[2];
  const float* w_gate = (const float*)d_in[3];
  const float* w_up   = (const float*)d_in[4];
  const float* w_down = (const float*)d_in[5];
  const float* sg     = (const float*)d_in[6];
  const float* su     = (const float*)d_in[7];
  const float* sd     = (const float*)d_in[8];
  float* out = (float*)d_out;

  char* ws = (char*)d_ws;
  size_t off = 0;
  auto take = [&](size_t bytes) -> void* {
    void* p = ws + off;
    off += (bytes + 255) & ~(size_t)255;
    return p;
  };
  half_t* h16        = (half_t*)take((size_t)kT * kH * 2);
  int*    topk_idx   = (int*)   take((size_t)kT * kK * 4);
  float*  topk_w     = (float*) take((size_t)kT * kK * 4);
  int*    counts     = (int*)   take(256);
  int*    slot_token = (int*)   take((size_t)kE * kCap * 4);
  int*    slot_pos   = (int*)   take((size_t)kT * kK * 4);
  half_t* inter_r    = (half_t*)take((size_t)kE * kCap * kI * 2);
  half_t* yb         = (half_t*)take((size_t)kE * kCap * kH * 2);
  half_t* inter_s    = (half_t*)take((size_t)kT * kI * 2);
  half_t* ysh        = (half_t*)take((size_t)kT * kH * 2);

  prep_kernel<<<(kT * kH) / 2048, 256, 0, stream>>>(x, h16, counts);
  router_kernel<<<kT / 8, 256, 0, stream>>>(x, gate_w, bias, topk_idx, topk_w);
  dispatch_kernel<<<(kT * kK) / 256, 256, 0, stream>>>(topk_idx, counts, slot_token, slot_pos);

  // routed experts
  moe_gateup_kernel<<<dim3(kCap / BM, kI / BN, kE), 256, 0, stream>>>(
      h16, slot_token, counts, w_gate, w_up, inter_r, kCap, kI, kH, 1e30f);
  moe_down_kernel<<<dim3(kCap / BM, kH / BN, kE), 256, 0, stream>>>(
      inter_r, counts, w_down, yb, kCap, kH, kI);

  // shared expert (identity rows, SwiGLU clamp at +/-10)
  moe_gateup_kernel<<<dim3(kT / BM, kI / BN, 1), 256, 0, stream>>>(
      h16, nullptr, nullptr, sg, su, inter_s, kT, kI, kH, kLimit);
  moe_down_kernel<<<dim3(kT / BM, kH / BN, 1), 256, 0, stream>>>(
      inter_s, nullptr, sd, ysh, kT, kH, kI);

  combine_kernel<<<kT, 256, 0, stream>>>(topk_idx, topk_w, slot_pos, yb, ysh, out);
}
